// AfmoeAttention_35227321761887
// MI455X (gfx1250) — matrix operand order repaired and checked by reading, not compiled, not run on hardware
//
#include <hip/hip_runtime.h>
#include <cstdint>
#include <cstddef>

// ---------------------------------------------------------------------------
// AFMoE attention block for MI455X (gfx1250, wave32, WMMA).
// All heavy math in v_wmma_f32_16x16x32_bf16; fp32 accumulate everywhere.
// d_in[3] (attention_mask) is unused: the sliding-window causal mask is
// applied analytically inside the flash-attention kernel.
// ---------------------------------------------------------------------------

typedef __attribute__((ext_vector_type(16))) __bf16 v16bf;
typedef __attribute__((ext_vector_type(8)))  __bf16 v8bf;
typedef __attribute__((ext_vector_type(4)))  __bf16 v4bf;
typedef __attribute__((ext_vector_type(8)))  float  v8f;
typedef __attribute__((ext_vector_type(4)))  float  v4f;

union BFrag { v16bf v; struct { v8bf lo, hi; } h; };

static constexpr int B_   = 2;
static constexpr int S_   = 2048;
static constexpr int H_   = 2048;
static constexpr int NH_  = 16;
static constexpr int NKV_ = 4;
static constexpr int D_   = 128;
static constexpr int GR_  = NH_ / NKV_;   // 4
static constexpr int WIN_ = 1024;
static constexpr float SCALE_ = 0.0883883476483184405f;  // 1/sqrt(128)

#define DEV __device__ __forceinline__

DEV v8f wmma_bf16(const BFrag& a, const BFrag& b, v8f c) {
  // D = A(16x32 bf16) * B(32x16 bf16) + C(16x16 f32)
  return __builtin_amdgcn_wmma_f32_16x16x32_bf16(
      /*neg_a=*/false, a.v, /*neg_b=*/false, b.v,
      /*c_mod=*/(short)0, c, /*reuse_a=*/false, /*reuse_b=*/false);
}

// ---------------------------------------------------------------------------
// fp32 -> bf16 conversion (vectorized x4)
// ---------------------------------------------------------------------------
__global__ void cvt_f32_bf16(const float* __restrict__ in,
                             __bf16* __restrict__ out, int n4) {
  int i = blockIdx.x * blockDim.x + threadIdx.x;
  if (i >= n4) return;
  v4f x = ((const v4f*)in)[i];
  v4bf y;
  y.x = (__bf16)x.x; y.y = (__bf16)x.y; y.z = (__bf16)x.z; y.w = (__bf16)x.w;
  ((v4bf*)out)[i] = y;
}

// ---------------------------------------------------------------------------
// C[M,N] (f32) = A[M,K] (bf16, row-major) * W[N,K]^T (bf16, row-major)
// Block: 256 threads = 8 waves; wave tile 32x64; block tile 128x128.
// Both operands are K-contiguous -> direct b128 fragment loads from L2.
// ---------------------------------------------------------------------------
__global__ __launch_bounds__(256)
void gemm_bf16_f32(const __bf16* __restrict__ A, const __bf16* __restrict__ Bw,
                   float* __restrict__ C, int M, int N, int K) {
  const int lane = threadIdx.x & 31;
  const int wv   = threadIdx.x >> 5;       // 0..7
  const int wm   = wv & 3;                 // 4 waves along M
  const int wn   = wv >> 2;                // 2 waves along N
  const int m0   = blockIdx.x * 128 + wm * 32;
  const int n0   = blockIdx.y * 128 + wn * 64;
  const int rt   = lane & 15;              // row (A) / column (B,C)
  const int hf   = lane >> 4;              // half selector

  const __bf16* a0 = A + (size_t)(m0 + rt) * K;
  const __bf16* a1 = A + (size_t)(m0 + 16 + rt) * K;

  v8f acc[2][4];
#pragma unroll
  for (int mt = 0; mt < 2; ++mt)
#pragma unroll
    for (int nt = 0; nt < 4; ++nt)
      acc[mt][nt] = (v8f){0.f,0.f,0.f,0.f,0.f,0.f,0.f,0.f};

  for (int k = 0; k < K; k += 32) {
    BFrag a[2];
    a[0].h.lo = *(const v8bf*)(a0 + k + hf * 16);
    a[0].h.hi = *(const v8bf*)(a0 + k + hf * 16 + 8);
    a[1].h.lo = *(const v8bf*)(a1 + k + hf * 16);
    a[1].h.hi = *(const v8bf*)(a1 + k + hf * 16 + 8);
#pragma unroll
    for (int nt = 0; nt < 4; ++nt) {
      const __bf16* bp = Bw + (size_t)(n0 + nt * 16 + rt) * K + k + hf * 16;
      BFrag bf;
      bf.h.lo = *(const v8bf*)bp;
      bf.h.hi = *(const v8bf*)(bp + 8);
      acc[0][nt] = wmma_bf16(a[0], bf, acc[0][nt]);
      acc[1][nt] = wmma_bf16(a[1], bf, acc[1][nt]);
    }
  }

#pragma unroll
  for (int mt = 0; mt < 2; ++mt)
#pragma unroll
    for (int nt = 0; nt < 4; ++nt)
#pragma unroll
      for (int r = 0; r < 8; ++r) {
        const int row = m0 + mt * 16 + 8 * hf + r;
        const int col = n0 + nt * 16 + rt;
        C[(size_t)row * N + col] = acc[mt][nt][r];
      }
}

// ---------------------------------------------------------------------------
// Fused RMSNorm + RoPE. One wave per head per token; lane owns 4 d-values,
// rotate_half partner (d +/- 64) fetched via __shfl_xor(16).
// Writes q_bf [b,h,s,d], k_bf [b,kv,s,d], and V transposed vT [b,kv,d,s]
// (so the P@V WMMA B-operand is K-contiguous).
// ---------------------------------------------------------------------------
DEV void rms_rope_store(const float* __restrict__ src, const float* __restrict__ w,
                        v4f c4, v4f s4, int lane, __bf16* __restrict__ dst) {
  v4f x = *(const v4f*)src;
  float ss = x.x * x.x + x.y * x.y + x.z * x.z + x.w * x.w;
#pragma unroll
  for (int m = 16; m >= 1; m >>= 1) ss += __shfl_xor(ss, m, 32);
  const float inv = rsqrtf(ss * (1.0f / (float)D_) + 1e-5f);
  v4f w4 = *(const v4f*)w;
  x = x * inv * w4;
  v4f o;
  o.x = __shfl_xor(x.x, 16, 32);
  o.y = __shfl_xor(x.y, 16, 32);
  o.z = __shfl_xor(x.z, 16, 32);
  o.w = __shfl_xor(x.w, 16, 32);
  const float sg = (lane < 16) ? -1.0f : 1.0f;  // rotate_half sign
  v4f r = x * c4 + o * (s4 * sg);
  v4bf y;
  y.x = (__bf16)r.x; y.y = (__bf16)r.y; y.z = (__bf16)r.z; y.w = (__bf16)r.w;
  *(v4bf*)dst = y;
}

__global__ __launch_bounds__(512)
void normrope_kernel(const float* __restrict__ qlin, const float* __restrict__ klin,
                     const float* __restrict__ vlin, const float* __restrict__ cosb,
                     const float* __restrict__ sinb, const float* __restrict__ qw,
                     const float* __restrict__ kw, __bf16* __restrict__ qbf,
                     __bf16* __restrict__ kbf, __bf16* __restrict__ vT) {
  const int tok  = blockIdx.x;             // b*S + s
  const int b    = tok / S_;
  const int s    = tok % S_;
  const int lane = threadIdx.x & 31;
  const int wv   = threadIdx.x >> 5;       // 0..15 -> q head id
  const int d4   = lane * 4;

  const v4f c4 = *(const v4f*)(cosb + ((size_t)b * S_ + s) * D_ + d4);
  const v4f s4 = *(const v4f*)(sinb + ((size_t)b * S_ + s) * D_ + d4);

  rms_rope_store(qlin + (size_t)tok * (NH_ * D_) + wv * D_ + d4, qw + d4,
                 c4, s4, lane,
                 qbf + (((size_t)b * NH_ + wv) * S_ + s) * D_ + d4);

  if (wv < NKV_) {
    rms_rope_store(klin + (size_t)tok * (NKV_ * D_) + wv * D_ + d4, kw + d4,
                   c4, s4, lane,
                   kbf + (((size_t)b * NKV_ + wv) * S_ + s) * D_ + d4);
  } else if (wv < 2 * NKV_) {
    const int kvh = wv - NKV_;
    v4f x = *(const v4f*)(vlin + (size_t)tok * (NKV_ * D_) + kvh * D_ + d4);
    __bf16* base = vT + ((size_t)b * NKV_ + kvh) * (size_t)D_ * S_ + s;
    base[(size_t)(d4 + 0) * S_] = (__bf16)x.x;
    base[(size_t)(d4 + 1) * S_] = (__bf16)x.y;
    base[(size_t)(d4 + 2) * S_] = (__bf16)x.z;
    base[(size_t)(d4 + 3) * S_] = (__bf16)x.w;
  }
}

// ---------------------------------------------------------------------------
// Flash attention, sliding-window causal, GQA, fused sigmoid-gate epilogue.
// Block = 128 threads = 4 waves; each wave owns 16 query rows (64 per block).
// Per k-block of 64 keys: 16 WMMA for QK^T, online softmax via LDS
// (same-wave DS ops are in-order -> no barriers; LDS partitioned per wave),
// 16 WMMA for P@V against pre-transposed V.
// ---------------------------------------------------------------------------
__global__ __launch_bounds__(128)
void attn_kernel(const __bf16* __restrict__ qbf, const __bf16* __restrict__ kbf,
                 const __bf16* __restrict__ vT, const float* __restrict__ gate,
                 __bf16* __restrict__ ao) {
  __shared__ __align__(16) float  sS[4][16][64];
  __shared__ __align__(16) __bf16 sP[4][16][64];
  __shared__ __align__(16) float  sA[4][16];
  __shared__ __align__(16) float  sL[4][16];

  const int lane = threadIdx.x & 31;
  const int w    = threadIdx.x >> 5;
  const int bh   = blockIdx.x;             // b*NH + h
  const int b    = bh / NH_;
  const int h    = bh % NH_;
  const int kvh  = h / GR_;
  const int q0   = blockIdx.y * 64 + w * 16;
  const int rt   = lane & 15;
  const int hf   = lane >> 4;

  const __bf16* Qp  = qbf + (((size_t)b * NH_ + h) * S_ + q0 + rt) * D_;
  const __bf16* Kp0 = kbf + ((size_t)b * NKV_ + kvh) * (size_t)S_ * D_;
  const __bf16* Vp0 = vT  + ((size_t)b * NKV_ + kvh) * (size_t)D_ * S_;

  BFrag qf[4];
#pragma unroll
  for (int j = 0; j < 4; ++j) {
    qf[j].h.lo = *(const v8bf*)(Qp + j * 32 + hf * 16);
    qf[j].h.hi = *(const v8bf*)(Qp + j * 32 + hf * 16 + 8);
  }

  v8f o[8];
#pragma unroll
  for (int t = 0; t < 8; ++t) o[t] = (v8f){0.f,0.f,0.f,0.f,0.f,0.f,0.f,0.f};

  float m_run = -3.0e38f, l_run = 0.0f;

  int kb0 = q0 - WIN_ + 1;
  if (kb0 < 0) kb0 = 0;
  kb0 &= ~63;

  for (int kb = kb0; kb <= q0 + 15; kb += 64) {
    // ---- scores S = Q K^T * scale + mask -> LDS -------------------------
#pragma unroll
    for (int t = 0; t < 4; ++t) {
      v8f acc = (v8f){0.f,0.f,0.f,0.f,0.f,0.f,0.f,0.f};
      const __bf16* kp = Kp0 + (size_t)(kb + t * 16 + rt) * D_;
#pragma unroll
      for (int j = 0; j < 4; ++j) {
        BFrag kf;
        kf.h.lo = *(const v8bf*)(kp + j * 32 + hf * 16);
        kf.h.hi = *(const v8bf*)(kp + j * 32 + hf * 16 + 8);
        acc = wmma_bf16(qf[j], kf, acc);
      }
      const int col = kb + t * 16 + rt;
#pragma unroll
      for (int r = 0; r < 8; ++r) {
        const int row = q0 + 8 * hf + r;
        float sc = acc[r] * SCALE_;
        sc += (col <= row && (row - col) < WIN_) ? 0.0f : -1.0e9f;
        sS[w][8 * hf + r][t * 16 + rt] = sc;
      }
    }
    // ---- online softmax: lanes 0..15 each own one row -------------------
    if (lane < 16) {
      const v4f* rp = (const v4f*)(&sS[w][lane][0]);
      float mb = -3.0e38f;
#pragma unroll
      for (int i = 0; i < 16; ++i) {
        v4f t = rp[i];
        mb = fmaxf(mb, fmaxf(fmaxf(t.x, t.y), fmaxf(t.z, t.w)));
      }
      const float nm = fmaxf(m_run, mb);
      const float al = __expf(m_run - nm);
      float sum = 0.0f;
#pragma unroll
      for (int i = 0; i < 16; ++i) {
        v4f t = rp[i];
        const float e0 = __expf(t.x - nm), e1 = __expf(t.y - nm);
        const float e2 = __expf(t.z - nm), e3 = __expf(t.w - nm);
        sum += (e0 + e1) + (e2 + e3);
        v4bf pb;
        pb.x = (__bf16)e0; pb.y = (__bf16)e1; pb.z = (__bf16)e2; pb.w = (__bf16)e3;
        *(v4bf*)(&sP[w][lane][i * 4]) = pb;
      }
      l_run = l_run * al + sum;
      m_run = nm;
      sA[w][lane] = al;
    }
    // ---- rescale O accumulators by alpha[row] ---------------------------
    {
      const v4f a0 = *(const v4f*)(&sA[w][8 * hf]);
      const v4f a1 = *(const v4f*)(&sA[w][8 * hf + 4]);
      const float alv[8] = {a0.x, a0.y, a0.z, a0.w, a1.x, a1.y, a1.z, a1.w};
#pragma unroll
      for (int t = 0; t < 8; ++t)
#pragma unroll
        for (int r = 0; r < 8; ++r) o[t][r] *= alv[r];
    }
    // ---- O += P @ V -----------------------------------------------------
#pragma unroll
    for (int ks = 0; ks < 2; ++ks) {
      BFrag pf;
      pf.h.lo = *(const v8bf*)(&sP[w][rt][ks * 32 + hf * 16]);
      pf.h.hi = *(const v8bf*)(&sP[w][rt][ks * 32 + hf * 16 + 8]);
#pragma unroll
      for (int dt = 0; dt < 8; ++dt) {
        const __bf16* vp = Vp0 + (size_t)(dt * 16 + rt) * S_ + kb + ks * 32 + hf * 16;
        BFrag vf;
        vf.h.lo = *(const v8bf*)vp;
        vf.h.hi = *(const v8bf*)(vp + 8);
        o[dt] = wmma_bf16(pf, vf, o[dt]);
      }
    }
  }

  // ---- epilogue: 1/l normalize, multiply by sigmoid(gate), write bf16 ----
  if (lane < 16) sL[w][lane] = 1.0f / l_run;
  const v4f l0 = *(const v4f*)(&sL[w][8 * hf]);
  const v4f l1 = *(const v4f*)(&sL[w][8 * hf + 4]);
  const float lv[8] = {l0.x, l0.y, l0.z, l0.w, l1.x, l1.y, l1.z, l1.w};
#pragma unroll
  for (int dt = 0; dt < 8; ++dt)
#pragma unroll
    for (int r = 0; r < 8; ++r) {
      const int row = q0 + 8 * hf + r;
      const int col = h * D_ + dt * 16 + rt;
      const size_t idx = ((size_t)b * S_ + row) * (size_t)(NH_ * D_) + col;
      const float g = gate[idx];
      const float sig = 1.0f / (1.0f + __expf(-g));
      ao[idx] = (__bf16)(o[dt][r] * lv[r] * sig);
    }
}

// ---------------------------------------------------------------------------
// Host-side orchestration
// ---------------------------------------------------------------------------
extern "C" void kernel_launch(void* const* d_in, const int* in_sizes, int n_in,
                              void* d_out, int out_size, void* d_ws, size_t ws_size,
                              hipStream_t stream) {
  (void)in_sizes; (void)n_in; (void)out_size; (void)ws_size;

  const float* x    = (const float*)d_in[0];   // hidden_states [B,S,H]
  const float* cosb = (const float*)d_in[1];   // [B,S,D]
  const float* sinb = (const float*)d_in[2];   // [B,S,D]
  /* d_in[3] attention_mask: mask computed analytically */
  const float* Wq   = (const float*)d_in[4];   // [NH*D, H]
  const float* Wk   = (const float*)d_in[5];   // [NKV*D, H]
  const float* Wv   = (const float*)d_in[6];   // [NKV*D, H]
  const float* Wo   = (const float*)d_in[7];   // [H, NH*D]
  const float* Wg   = (const float*)d_in[8];   // [NH*D, H]
  const float* qw   = (const float*)d_in[9];   // [D]
  const float* kw   = (const float*)d_in[10];  // [D]
  float* out = (float*)d_out;                  // [B,S,H] fp32

  const size_t M   = (size_t)B_ * S_;          // 4096 tokens
  const size_t nX  = M * H_;                   // 8388608
  const size_t nWq = (size_t)NH_ * D_ * H_;    // 4194304
  const size_t nWk = (size_t)NKV_ * D_ * H_;   // 1048576

  // Workspace carve (256B aligned).
  char* p = (char*)d_ws;
  auto take = [&p](size_t bytes) {
    char* r = p;
    p += (bytes + 255) & ~(size_t)255;
    return r;
  };
  __bf16* xb   = (__bf16*)take(nX  * 2);
  __bf16* Wqb  = (__bf16*)take(nWq * 2);
  __bf16* Wkb  = (__bf16*)take(nWk * 2);
  __bf16* Wvb  = (__bf16*)take(nWk * 2);
  __bf16* Wgb  = (__bf16*)take(nWq * 2);
  __bf16* Wob  = (__bf16*)take(nWq * 2);
  float*  qlin = (float*) take(nX  * 4);       // reused as `ao` region later
  float*  klin = (float*) take(M * NKV_ * D_ * 4);
  float*  vlin = (float*) take(M * NKV_ * D_ * 4);
  float*  gbuf = (float*) take(nX  * 4);
  __bf16* qbf  = (__bf16*)take(nX  * 2);
  __bf16* kbf  = (__bf16*)take(M * NKV_ * D_ * 2);
  __bf16* vTb  = (__bf16*)take(M * NKV_ * D_ * 2);
  __bf16* ao   = (__bf16*)qlin;                // alias: qlin dead after normrope

  // 1) fp32 -> bf16 conversions
  auto cvt = [&](const float* src, __bf16* dst, size_t n) {
    int n4 = (int)(n / 4);
    cvt_f32_bf16<<<(n4 + 255) / 256, 256, 0, stream>>>(src, dst, n4);
  };
  cvt(x,  xb,  nX);
  cvt(Wq, Wqb, nWq);
  cvt(Wk, Wkb, nWk);
  cvt(Wv, Wvb, nWk);
  cvt(Wg, Wgb, nWq);
  cvt(Wo, Wob, nWq);

  // 2) projections: y = x @ W^T   (M x N x K)
  gemm_bf16_f32<<<dim3(4096 / 128, 2048 / 128), 256, 0, stream>>>(
      xb, Wqb, qlin, 4096, 2048, 2048);
  gemm_bf16_f32<<<dim3(4096 / 128,  512 / 128), 256, 0, stream>>>(
      xb, Wkb, klin, 4096,  512, 2048);
  gemm_bf16_f32<<<dim3(4096 / 128,  512 / 128), 256, 0, stream>>>(
      xb, Wvb, vlin, 4096,  512, 2048);
  gemm_bf16_f32<<<dim3(4096 / 128, 2048 / 128), 256, 0, stream>>>(
      xb, Wgb, gbuf, 4096, 2048, 2048);

  // 3) RMSNorm + RoPE + V transpose
  normrope_kernel<<<B_ * S_, 512, 0, stream>>>(
      qlin, klin, vlin, cosb, sinb, qw, kw, qbf, kbf, vTb);

  // 4) flash attention + sigmoid gate
  attn_kernel<<<dim3(B_ * NH_, S_ / 64), 128, 0, stream>>>(
      qbf, kbf, vTb, gbuf, ao);

  // 5) output projection: out = gated_attn @ Wo^T
  gemm_bf16_f32<<<dim3(4096 / 128, 2048 / 128), 256, 0, stream>>>(
      ao, Wob, out, 4096, 2048, 2048);
}
